// ACE_41094247088736
// MI455X (gfx1250) — matrix operand order repaired and checked by reading, not compiled, not run on hardware
//
#include <hip/hip_runtime.h>
#include <hip/hip_bf16.h>
#include <stdint.h>

// Problem constants (match reference)
#define BB 4
#define CC 128
#define HH 128
#define WW 128
#define LL 19
#define SS 512
#define HP 132          // H + 4 (halo of 2 each side)
#define WP 132          // W + 4
#define KCH 1024        // concatenated channels: 512 middle_avg + 512 actv
#define KTOT 25600      // 25 taps * 1024 channels
#define NOC 256         // 128 gamma + 128 beta output channels

typedef _Float16 v16h __attribute__((ext_vector_type(16)));
typedef _Float16 half8 __attribute__((ext_vector_type(8)));
typedef float v8f __attribute__((ext_vector_type(8)));

union AFrag { v16h v; half8 h[2]; };

// CDNA5 async global->LDS copy (ASYNCcnt-tracked; ISA §15.18.3 opcode 98).
// ldsoff: wave-relative LDS byte address (low 32 bits of generic LDS pointer).
__device__ __forceinline__ void async_cp16(uint32_t ldsoff, const _Float16* g) {
    asm volatile("global_load_async_to_lds_b128 %0, %1, off"
                 :: "v"(ldsoff), "v"(g)
                 : "memory");
}

// ---------------------------------------------------------------------------
// K0: zero the padded NHWC f16 tensor (halo must be zero for branch-free conv)
__global__ void k0_zero(uint4* __restrict__ p, size_t n16) {
    size_t i = (size_t)blockIdx.x * blockDim.x + threadIdx.x;
    size_t stride = (size_t)gridDim.x * blockDim.x;
    uint4 z; z.x = z.y = z.z = z.w = 0u;
    for (; i < n16; i += stride) p[i] = z;
}

// ---------------------------------------------------------------------------
// K1: instance-norm statistics, one block per (b,c), reduce 128*128 elems
__global__ void k1_stats(const float* __restrict__ x,
                         float* __restrict__ mean, float* __restrict__ rstd) {
    __shared__ float s1[256];
    __shared__ float s2[256];
    int bc = blockIdx.x;                     // 0..511
    const float* p = x + (size_t)bc * (HH * WW);
    float s = 0.f, ss = 0.f;
    for (int i = threadIdx.x; i < HH * WW; i += 256) {
        float v = p[i]; s += v; ss += v * v;
    }
    s1[threadIdx.x] = s; s2[threadIdx.x] = ss;
    __syncthreads();
    for (int o = 128; o > 0; o >>= 1) {
        if (threadIdx.x < (unsigned)o) {
            s1[threadIdx.x] += s1[threadIdx.x + o];
            s2[threadIdx.x] += s2[threadIdx.x + o];
        }
        __syncthreads();
    }
    if (threadIdx.x == 0) {
        float m = s1[0] * (1.f / (HH * WW));
        float var = s2[0] * (1.f / (HH * WW)) - m * m;
        mean[bc] = m;
        rstd[bc] = rsqrtf(var + 1e-5f);
    }
}

// ---------------------------------------------------------------------------
// K2: mu[b,j,e] = relu(style_codes[b,j,:] . fc_w[j,e,:] + fc_b[j,e])
__global__ void k2_stylefc(const float* __restrict__ sc,
                           const float* __restrict__ fcw,
                           const float* __restrict__ fcb,
                           float* __restrict__ mu) {
    __shared__ float sv[SS];
    int bj = blockIdx.x;                     // 0..B*L-1
    int j = bj % LL;
    const float* s = sc + (size_t)bj * SS;
    for (int i = threadIdx.x; i < SS; i += 256) sv[i] = s[i];
    __syncthreads();
    for (int e = threadIdx.x; e < SS; e += 256) {
        const float* wr = fcw + ((size_t)j * SS + e) * SS;
        float acc = fcb[j * SS + e];
        for (int d = 0; d < SS; ++d) acc += sv[d] * wr[d];
        mu[(size_t)bj * SS + e] = fmaxf(acc, 0.f);
    }
}

// ---------------------------------------------------------------------------
// K3: recover segment id per pixel from the one-hot segmap (argmax over L)
__global__ void k3_segid(const float* __restrict__ segmap, int* __restrict__ sid) {
    int idx = blockIdx.x * 256 + threadIdx.x;        // 0..B*H*W-1
    if (idx >= BB * HH * WW) return;
    int b = idx >> 14;
    int hw = idx & (HH * WW - 1);
    const float* p = segmap + (size_t)b * LL * (HH * WW) + hw;
    int best = 0; float bv = p[0];
    for (int j = 1; j < LL; ++j) {
        float v = p[(size_t)j * (HH * WW)];
        if (v > bv) { bv = v; best = j; }
    }
    sid[idx] = best;
}

// ---------------------------------------------------------------------------
// K4: middle_avg scatter -> padded NHWC f16 tensor X2[:, :, :, 0:512]
__global__ void k4_scatter(const float* __restrict__ mu, const int* __restrict__ sid,
                           _Float16* __restrict__ X2) {
    size_t idx = (size_t)blockIdx.x * 256 + threadIdx.x;   // B*H*W*512
    if (idx >= (size_t)BB * HH * WW * SS) return;
    int e = (int)(idx & (SS - 1));
    size_t p = idx >> 9;
    int b = (int)(p >> 14);
    int hw = (int)(p & (HH * WW - 1));
    int h = hw >> 7, w = hw & (WW - 1);
    int s = sid[p];
    float v = mu[((size_t)(b * LL + s)) * SS + e];
    X2[(((size_t)(b * HP) + h + 2) * WP + (w + 2)) * KCH + e] = (_Float16)v;
}

// ---------------------------------------------------------------------------
// K5: actv = relu(conv5(one-hot segmap, spade_shared_w) + b)
//     one-hot conv == 25 weight-row lookups. Writes X2[:, :, :, 512:1024].
__global__ void k5_sharedconv(const int* __restrict__ sid,
                              const float* __restrict__ Wsh,   // [512][19][25]
                              const float* __restrict__ bsh,   // [512]
                              _Float16* __restrict__ X2) {
    __shared__ int ssid[25];
    int p = blockIdx.x;                       // pixel 0..B*H*W-1
    int b = p >> 14;
    int hw = p & (HH * WW - 1);
    int h = hw >> 7, w = hw & (WW - 1);
    if (threadIdx.x < 25) {
        int kh = threadIdx.x / 5, kw = threadIdx.x % 5;
        int hh = h + kh - 2, ww = w + kw - 2;
        ssid[threadIdx.x] = (hh >= 0 && hh < HH && ww >= 0 && ww < WW)
                          ? sid[(b << 14) + (hh << 7) + ww] : -1;
    }
    __syncthreads();
    size_t obase = (((size_t)(b * HP) + h + 2) * WP + (w + 2)) * KCH + SS;
    for (int e = threadIdx.x; e < SS; e += 256) {
        float acc = bsh[e];
        const float* wr = Wsh + (size_t)e * (LL * 25);
#pragma unroll
        for (int t = 0; t < 25; ++t) {
            int s = ssid[t];
            if (s >= 0) acc += wr[s * 25 + t];
        }
        X2[obase + e] = (_Float16)fmaxf(acc, 0.f);
    }
}

// ---------------------------------------------------------------------------
// K6: build concatenated, sigmoid-blended f16 weight matrix Wcat[256][25600]
__global__ void k6_prepw(const float* __restrict__ cg, const float* __restrict__ cb,
                         const float* __restrict__ sg, const float* __restrict__ sb,
                         const float* __restrict__ cgb, const float* __restrict__ cbb,
                         const float* __restrict__ sgb, const float* __restrict__ sbb,
                         const float* __restrict__ blg, const float* __restrict__ blb,
                         _Float16* __restrict__ Wcat, float* __restrict__ biasN) {
    float ga = 1.f / (1.f + expf(-blg[0]));
    float ba = 1.f / (1.f + expf(-blb[0]));
    int tid0 = blockIdx.x * 256 + threadIdx.x;
    if (tid0 < NOC) {
        int oc = tid0, ocm = oc & 127;
        biasN[oc] = (oc < 128) ? (ga * cgb[ocm] + (1.f - ga) * sgb[ocm])
                               : (ba * cbb[ocm] + (1.f - ba) * sbb[ocm]);
    }
    size_t total = (size_t)NOC * KTOT;
    size_t stride = (size_t)gridDim.x * 256;
    for (size_t idx = tid0; idx < total; idx += stride) {
        int oc = (int)(idx / KTOT);
        int k = (int)(idx % KTOT);
        int tap = k >> 10;
        int ic = k & (KCH - 1);
        int ocm = oc & 127;
        bool isBeta = oc >= 128;
        float a = isBeta ? ba : ga;
        float v;
        if (ic < SS) {
            const float* Wv = isBeta ? cb : cg;          // [128][512][25]
            v = a * Wv[((size_t)(ocm * SS + ic)) * 25 + tap];
        } else {
            const float* Wv = isBeta ? sb : sg;
            v = (1.f - a) * Wv[((size_t)(ocm * SS + (ic - SS))) * 25 + tap];
        }
        Wcat[idx] = (_Float16)v;
    }
}

// ---------------------------------------------------------------------------
// K7: fused implicit-GEMM conv (WMMA f32_16x16x32_f16) + instance-norm epilogue.
//     Per tap, the block's 16-pixel x 1024-ch A tile (one contiguous 32 KB
//     region of padded NHWC) is async-copied to LDS (double-buffered, XOR
//     bank swizzle) and shared by all 8 waves; B (weights) streams from L2.
__global__ void __launch_bounds__(256)
k7_convwmma(const _Float16* __restrict__ X2,     // [B][HP][WP][1024] f16, padded
            const _Float16* __restrict__ Wcat,   // [256][25600] f16
            const float* __restrict__ biasN,     // [256]
            const float* __restrict__ x,         // [B][C][H][W] f32
            const float* __restrict__ mean,      // [B*C]
            const float* __restrict__ rstd,      // [B*C]
            float* __restrict__ out) {           // [B][C][H][W] f32
    __shared__ _Float16 abuf[2][16 * KCH];       // 2 x 32 KB double buffer

    int tile = blockIdx.x;                 // 0..4095 = B * H * (W/16)
    int b  = tile >> 10;                   // 1024 tiles per image (128 rows * 8)
    int h  = (tile >> 3) & (HH - 1);
    int w0 = (tile & 7) << 4;

    int wave = threadIdx.x >> 5;           // 0..7  (wave32)
    int lane = threadIdx.x & 31;
    int half = lane >> 4;
    int lm   = lane & 15;                  // M index for A, N index for B/C/D
    int oc0  = wave << 4;
    int sw   = lm << 4;                    // per-pixel XOR bank swizzle

    // Wave-relative LDS byte offsets of the two buffers (generic addr low 32b)
    uint32_t lds0 = (uint32_t)(uintptr_t)&abuf[0][0];
    uint32_t lds1 = (uint32_t)(uintptr_t)&abuf[1][0];

    // Issue the 32 KB tap tile copy: 2048 16-B chunks, 8 per thread.
    // LDS layout: pixel-major 2048 B/pixel; within-offset XORed by pixel*16.
    auto issue_copy = [&](int tap, uint32_t ldsb) {
        int kh = tap / 5, kw = tap % 5;
        const _Float16* gsrc =
            X2 + (((size_t)(b * HP) + h + kh) * WP + (w0 + kw)) * KCH;
#pragma unroll
        for (int i = 0; i < 8; ++i) {
            int c16 = (int)threadIdx.x + (i << 8);   // 0..2047
            int pix = c16 >> 7;                      // 128 chunks per pixel
            int within = (c16 & 127) << 4;           // byte offset in pixel
            uint32_t dst = ldsb + (uint32_t)((pix << 11) + (within ^ (pix << 4)));
            async_cp16(dst, gsrc + ((size_t)c16 << 3));
        }
    };

    issue_copy(0, lds0);

    v8f accG = {};
    v8f accB = {};

    // B-fragment row bases: lane holds 16 consecutive K at offset 16*half
    const _Float16* wg = Wcat + (size_t)(oc0 + lm) * KTOT + 16 * half;
    const _Float16* wb = Wcat + (size_t)(128 + oc0 + lm) * KTOT + 16 * half;

    for (int tap = 0; tap < 25; ++tap) {
        if (tap + 1 < 25) {
            issue_copy(tap + 1, (tap & 1) ? lds0 : lds1);
            asm volatile("s_wait_asynccnt 0x8" ::: "memory");  // tap's copy done
        } else {
            asm volatile("s_wait_asynccnt 0x0" ::: "memory");
        }
        __syncthreads();   // all waves' copies for this tap visible

        const char* abase = (const char*)&abuf[0][0]
                          + ((tap & 1) ? (16 * KCH * 2) : 0) + (lm << 11);
        const _Float16* wgt = wg + tap * KCH;
        const _Float16* wbt = wb + tap * KCH;
        for (int ic0 = 0; ic0 < KCH; ic0 += 32) {
            int W1 = (ic0 + 16 * half) * 2;          // byte offset of K chunk
            AFrag a;
            a.h[0] = *(const half8*)(abase + (W1 ^ sw));
            a.h[1] = *(const half8*)(abase + ((W1 + 16) ^ sw));
            v16h bg  = *(const v16h*)(wgt + ic0);
            v16h bbv = *(const v16h*)(wbt + ic0);
            accG = __builtin_amdgcn_wmma_f32_16x16x32_f16(
                false, a.v, false, bg, (short)0, accG, false, false);
            accB = __builtin_amdgcn_wmma_f32_16x16x32_f16(
                false, a.v, false, bbv, (short)0, accB, false, false);
        }
        __syncthreads();   // reads done before next iteration overwrites buffer
    }

    // Epilogue: D layout -> lane = out channel, VGPR v = pixel (v + 8*half).
    int c = oc0 + lm;
    float mn = mean[b * CC + c];
    float rs = rstd[b * CC + c];
    float bg0 = biasN[c];
    float bb0 = biasN[128 + c];
#pragma unroll
    for (int v = 0; v < 8; ++v) {
        int m = v + 8 * half;
        int w = w0 + m;
        size_t xi = (((size_t)(b * CC + c)) * HH + h) * WW + w;
        float xn = (x[xi] - mn) * rs;
        float g  = accG[v] + bg0;
        float be = accB[v] + bb0;
        out[xi] = xn * (1.f + g) + be;
    }
}

// ---------------------------------------------------------------------------
static inline size_t alignup(size_t v) { return (v + 255) & ~(size_t)255; }

extern "C" void kernel_launch(void* const* d_in, const int* in_sizes, int n_in,
                              void* d_out, int out_size, void* d_ws, size_t ws_size,
                              hipStream_t stream) {
    (void)in_sizes; (void)n_in; (void)out_size; (void)ws_size;

    const float* x      = (const float*)d_in[0];
    const float* segmap = (const float*)d_in[1];
    const float* sc     = (const float*)d_in[2];
    const float* fcw    = (const float*)d_in[3];
    const float* fcb    = (const float*)d_in[4];
    const float* cgw    = (const float*)d_in[5];
    const float* cgb    = (const float*)d_in[6];
    const float* cbw    = (const float*)d_in[7];
    const float* cbb    = (const float*)d_in[8];
    const float* ssw    = (const float*)d_in[9];
    const float* ssb    = (const float*)d_in[10];
    const float* sgw    = (const float*)d_in[11];
    const float* sgb    = (const float*)d_in[12];
    const float* sbw    = (const float*)d_in[13];
    const float* sbb    = (const float*)d_in[14];
    const float* blg    = (const float*)d_in[15];
    const float* blb    = (const float*)d_in[16];
    float* out = (float*)d_out;

    // Workspace carve-up
    char* w = (char*)d_ws;
    float* mean = (float*)w;               w += alignup(sizeof(float) * BB * CC);
    float* rstd = (float*)w;               w += alignup(sizeof(float) * BB * CC);
    float* mu   = (float*)w;               w += alignup(sizeof(float) * BB * LL * SS);
    int*   sid  = (int*)w;                 w += alignup(sizeof(int) * BB * HH * WW);
    float* biasN = (float*)w;              w += alignup(sizeof(float) * NOC);
    _Float16* Wcat = (_Float16*)w;         w += alignup(sizeof(_Float16) * (size_t)NOC * KTOT);
    _Float16* X2 = (_Float16*)w;           // padded NHWC: B*HP*WP*KCH f16
    size_t x2_bytes = sizeof(_Float16) * (size_t)BB * HP * WP * KCH;

    // K0: zero padded tensor (halo zeros make the conv branch-free)
    {
        size_t n16 = x2_bytes / 16;
        int blocks = (int)((n16 + 255) / 256);
        k0_zero<<<blocks, 256, 0, stream>>>((uint4*)X2, n16);
    }
    // K1: instance-norm stats
    k1_stats<<<BB * CC, 256, 0, stream>>>(x, mean, rstd);
    // K2: per-region style FC
    k2_stylefc<<<BB * LL, 256, 0, stream>>>(sc, fcw, fcb, mu);
    // K3: one-hot -> segment id
    k3_segid<<<(BB * HH * WW) / 256, 256, 0, stream>>>(segmap, sid);
    // K4: scatter mu into X2[:, :, :, 0:512]
    {
        size_t total = (size_t)BB * HH * WW * SS;
        k4_scatter<<<(int)(total / 256), 256, 0, stream>>>(mu, sid, X2);
    }
    // K5: shared conv (one-hot trick) into X2[:, :, :, 512:1024]
    k5_sharedconv<<<BB * HH * WW, 256, 0, stream>>>(sid, ssw, ssb, X2);
    // K6: blended concatenated weights + bias
    k6_prepw<<<25600, 256, 0, stream>>>(cgw, cbw, sgw, sbw,
                                        cgb, cbb, sgb, sbb,
                                        blg, blb, Wcat, biasN);
    // K7: fused WMMA implicit GEMM + norm/blend epilogue
    k7_convwmma<<<BB * HH * (WW / 16), 256, 0, stream>>>(
        X2, Wcat, biasN, x, mean, rstd, out);
}
